// ParallelContextualAttention_4922032521652
// MI455X (gfx1250) — matrix operand order repaired and checked by reading, not compiled, not run on hardware
//
#include <hip/hip_runtime.h>
#include <hip/hip_bf16.h>

// ---------------------------------------------------------------------------
// ParallelContextualAttention for MI455X (gfx1250, wave32, WMMA bf16,
// async global->LDS staging with double buffering).
//
// Pipeline per (sample b, branch p in {1,3}):
//   build_F:   F[s,k]      = fg patch (zero-padded), bf16            (4096 x K)
//   build_A:   A[l,k]      = (bg patch + 1e-7)/||.||, bf16           (4096 x K)
//   gemm1:     ST[s,l]     = sum_k F[s,k]*A[l,k]          (f32, 4096 x 4096)
//   box+smax:  ATT[s,l]    = softmax_l( 3x3-boxsum_s ST ) (bf16)
//   gemm2:     U[s,k]      = sum_l ATT[s,l]*A[l,k]        (f32, 4096 x K)
//   col2im:    outs[c,s]   = col2im(U)*m/p^2 + fg*(1-m)
// Epilogue: channel means -> SE gate -> 1x1 combiner.
// K = 64*p*p (64 or 576).  ST (64MB) fits in the 192MB L2, so the 9x
// neighbor reads in box+softmax stay on-chip.
// ---------------------------------------------------------------------------

typedef __attribute__((ext_vector_type(16))) __bf16          v16bf;
typedef __attribute__((ext_vector_type(16))) unsigned short  v16us;
typedef __attribute__((ext_vector_type(8)))  float           v8f;

#define LSP   4096      // number of spatial locations = number of patches
#define NCH   64        // channels
#define KMAX  576       // max patch dim (64*9)
#define LDA   40        // LDS row stride (ushorts) for 32-wide K tiles (80B, 16B-aligned)
#define LDB   72        // LDS row stride (ushorts) for 64-wide N tiles (144B, 16B-aligned)

__device__ __forceinline__ unsigned short f2bf(float x) {
  union { float f; unsigned int u; } v; v.f = x;
  unsigned int r = v.u + 0x7FFFu + ((v.u >> 16) & 1u);   // round-to-nearest-even
  return (unsigned short)(r >> 16);
}

// ---- CDNA5 async global->LDS copy (ASYNCcnt-tracked, no VGPR round-trip) ---
// LDS generic pointers: the shared aperture occupies a full 4GB window
// (isLDS checks addr[63:32] only), so the low 32 bits ARE the LDS offset.

__device__ __forceinline__ void async_b128(const unsigned short* lds_dst,
                                           const unsigned short* gsrc) {
  const unsigned d = (unsigned)(unsigned long long)lds_dst;
  asm volatile("global_load_async_to_lds_b128 %0, %1, off"
               :: "v"(d), "v"(gsrc) : "memory");
}
#define WAIT_ASYNC(imm) asm volatile("s_wait_asynccnt " #imm ::: "memory")

// ---- WMMA fragment gathers (contiguous k = 16*half + s in A and B, wave32) --------------

// A-matrix 16x32 bf16: lane m in [0,16) rows, halves split K.
__device__ __forceinline__ v16bf frag_a(const unsigned short* t, int row0, int ld, int lane) {
  const int half = lane >> 4, m = lane & 15;
  const unsigned short* p = t + (size_t)(row0 + m) * ld;
  v16us u;
#pragma unroll
  for (int tt = 0; tt < 8; ++tt) {
    const int kb = (half << 4) + (tt << 1);
    u[2 * tt]     = p[kb];
    u[2 * tt + 1] = p[kb + 1];
  }
  return __builtin_bit_cast(v16bf, u);
}

// B-matrix 32x16 bf16 from LDS laid out as [k][n] rows.
__device__ __forceinline__ v16bf frag_b_cols(const unsigned short* t, int col0, int ld, int lane) {
  const int half = lane >> 4, n = lane & 15;
  v16us u;
#pragma unroll
  for (int tt = 0; tt < 8; ++tt) {
    const int k = (half << 4) + (tt << 1);
    u[2 * tt]     = t[(size_t)k * ld + col0 + n];
    u[2 * tt + 1] = t[(size_t)(k + 1) * ld + col0 + n];
  }
  return __builtin_bit_cast(v16bf, u);
}

// B-matrix 32x16 bf16 where LDS holds one row per N element ([n][k] rows).
__device__ __forceinline__ v16bf frag_b_rows(const unsigned short* t, int nrow0, int ld, int lane) {
  const int half = lane >> 4, n = lane & 15;
  const unsigned short* p = t + (size_t)(nrow0 + n) * ld + (half << 4);
  v16us u;
#pragma unroll
  for (int tt = 0; tt < 8; ++tt) {
    u[2 * tt]     = p[2 * tt];
    u[2 * tt + 1] = p[2 * tt + 1];
  }
  return __builtin_bit_cast(v16bf, u);
}

// C/D 16x16 f32: VGPR r -> row (r + 8*half), col = lane&15.
__device__ __forceinline__ void store_frag(float* __restrict__ D, size_t ld,
                                           int row0, int col0, int lane, v8f a) {
  const int half = lane >> 4, n = lane & 15;
#pragma unroll
  for (int r = 0; r < 8; ++r)
    D[(size_t)(row0 + (half << 3) + r) * ld + col0 + n] = a[r];
}

// ---- patch builders --------------------------------------------------------

__global__ __launch_bounds__(256) void build_F_k(const float* __restrict__ fg,
                                                 unsigned short* __restrict__ Fb,
                                                 int p, int Kdim) {
  const int idx = blockIdx.x * 256 + threadIdx.x;
  if (idx >= LSP * Kdim) return;
  const int s = idx / Kdim, k = idx - s * Kdim;
  float val;
  if (p == 1) {
    val = fg[(size_t)k * LSP + s];
  } else {
    const int c = k / 9, ij = k - c * 9, i = ij / 3, j = ij - i * 3;
    const int y = s >> 6, x = s & 63;
    const int yy = y + i - 1, xx = x + j - 1;
    val = (yy >= 0 && yy < 64 && xx >= 0 && xx < 64)
              ? fg[(size_t)c * LSP + yy * 64 + xx] : 0.0f;
  }
  Fb[idx] = f2bf(val);
}

__global__ __launch_bounds__(256) void build_A_k(const float* __restrict__ fg,
                                                 const float* __restrict__ mask,
                                                 unsigned short* __restrict__ Ab,
                                                 int p, int Kdim) {
  __shared__ float vals[KMAX];
  __shared__ float red[256];
  const int l = blockIdx.x, tid = threadIdx.x;
  const int y = l >> 6, x = l & 63;
  float ss = 0.0f;
  for (int k = tid; k < Kdim; k += 256) {
    float v;
    if (p == 1) {
      v = fg[(size_t)k * LSP + l] * (1.0f - mask[l]) + 1e-7f;
    } else {
      const int c = k / 9, ij = k - c * 9, i = ij / 3, j = ij - i * 3;
      const int yy = y + i - 1, xx = x + j - 1;
      if (yy >= 0 && yy < 64 && xx >= 0 && xx < 64) {
        const int sn = yy * 64 + xx;
        v = fg[(size_t)c * LSP + sn] * (1.0f - mask[sn]) + 1e-7f;
      } else {
        v = 1e-7f;                 // pad entries: 0 + 1e-7, included in the norm
      }
    }
    vals[k] = v;
    ss += v * v;
  }
  red[tid] = ss; __syncthreads();
  for (int st = 128; st > 0; st >>= 1) {
    if (tid < st) red[tid] += red[tid + st];
    __syncthreads();
  }
  const float inv = rsqrtf(red[0]);
  for (int k = tid; k < Kdim; k += 256)
    Ab[(size_t)l * Kdim + k] = f2bf(vals[k] * inv);
}

// ---- GEMM1: ST[s,l] = sum_k F[s,k] * A[l,k]  (M=N=4096, K=Kdim) ------------
// 256 threads = 8 wave32, 128x128 tile, wave -> 32x64 (2x4 16x16 frags).
// Double-buffered async global->LDS staging (4 async b128 per thread/tile).

__global__ __launch_bounds__(256) void gemm_scores_k(const unsigned short* __restrict__ Fb,
                                                     const unsigned short* __restrict__ Ab,
                                                     float* __restrict__ ST, int Kdim) {
  __shared__ unsigned short sF[2][128 * LDA];   // [m][k], padded rows
  __shared__ unsigned short sA[2][128 * LDA];   // [n(l)][k], padded rows
  const int tid = threadIdx.x, lane = tid & 31, wave = tid >> 5;
  const int wm = wave >> 1, wn = wave & 1;
  const int m0 = blockIdx.x * 128;              // s tile
  const int n0 = blockIdx.y * 128;              // l tile

  const int r  = tid >> 1;                      // 0..127: tile row this thread stages
  const int hh = (tid & 1) << 4;                // 0 or 16 ushorts (32B halves)
  const unsigned short* gF = Fb + (size_t)(m0 + r) * Kdim + hh;
  const unsigned short* gA = Ab + (size_t)(n0 + r) * Kdim + hh;
  const int rowoff = r * LDA + hh;

  v8f acc[2][4];
#pragma unroll
  for (int mi = 0; mi < 2; ++mi)
#pragma unroll
    for (int ni = 0; ni < 4; ++ni) acc[mi][ni] = {};

  // prologue: stage k-tile 0 into buffer 0
  async_b128(&sF[0][rowoff],     gF);
  async_b128(&sF[0][rowoff + 8], gF + 8);
  async_b128(&sA[0][rowoff],     gA);
  async_b128(&sA[0][rowoff + 8], gA + 8);

  const int nk = Kdim >> 5;
  for (int kt = 0; kt < nk; ++kt) {
    const int cur = kt & 1, nxt = cur ^ 1;
    if (kt + 1 < nk) {                          // stream next tile while computing
      const unsigned short* nF = gF + (size_t)(kt + 1) * 32;
      const unsigned short* nA = gA + (size_t)(kt + 1) * 32;
      async_b128(&sF[nxt][rowoff],     nF);
      async_b128(&sF[nxt][rowoff + 8], nF + 8);
      async_b128(&sA[nxt][rowoff],     nA);
      async_b128(&sA[nxt][rowoff + 8], nA + 8);
      WAIT_ASYNC(0x4);                          // tile kt done; tile kt+1 in flight
    } else {
      WAIT_ASYNC(0x0);
    }
    __syncthreads();

    v16bf af[2], bfm[4];
#pragma unroll
    for (int mi = 0; mi < 2; ++mi) af[mi] = frag_a(sF[cur], wm * 32 + mi * 16, LDA, lane);
#pragma unroll
    for (int ni = 0; ni < 4; ++ni) bfm[ni] = frag_b_rows(sA[cur], wn * 64 + ni * 16, LDA, lane);
#pragma unroll
    for (int mi = 0; mi < 2; ++mi)
#pragma unroll
      for (int ni = 0; ni < 4; ++ni)
        acc[mi][ni] = __builtin_amdgcn_wmma_f32_16x16x32_bf16(
            false, af[mi], false, bfm[ni], (short)0, acc[mi][ni], false, false);
    __syncthreads();
  }
#pragma unroll
  for (int mi = 0; mi < 2; ++mi)
#pragma unroll
    for (int ni = 0; ni < 4; ++ni)
      store_frag(ST, LSP, m0 + wm * 32 + mi * 16, n0 + wn * 64 + ni * 16, lane, acc[mi][ni]);
}

// ---- fused 3x3 box-sum (over query spatial dims) + softmax over l ----------

__global__ __launch_bounds__(256) void box_softmax_k(const float* __restrict__ ST,
                                                     unsigned short* __restrict__ ATT) {
  __shared__ float red[256];
  const int s = blockIdx.x, tid = threadIdx.x;
  const int y = s >> 6, x = s & 63;
  float v[16];
  float lmax = -3.4e38f;
#pragma unroll
  for (int kk = 0; kk < 16; ++kk) {
    const int l = kk * 256 + tid;
    float sum = 0.0f;
#pragma unroll
    for (int dy = -1; dy <= 1; ++dy) {
      const int yy = y + dy;
      if (yy < 0 || yy >= 64) continue;
#pragma unroll
      for (int dx = -1; dx <= 1; ++dx) {
        const int xx = x + dx;
        if (xx < 0 || xx >= 64) continue;
        sum += ST[(size_t)(yy * 64 + xx) * LSP + l];   // L2-resident rows
      }
    }
    v[kk] = sum;
    lmax = fmaxf(lmax, sum);
  }
  red[tid] = lmax; __syncthreads();
  for (int st = 128; st > 0; st >>= 1) {
    if (tid < st) red[tid] = fmaxf(red[tid], red[tid + st]);
    __syncthreads();
  }
  const float M = red[0];
  __syncthreads();
  float lsum = 0.0f;
#pragma unroll
  for (int kk = 0; kk < 16; ++kk) {
    const float e = __expf(v[kk] - M);
    v[kk] = e;
    lsum += e;
  }
  red[tid] = lsum; __syncthreads();
  for (int st = 128; st > 0; st >>= 1) {
    if (tid < st) red[tid] += red[tid + st];
    __syncthreads();
  }
  const float inv = 1.0f / red[0];
#pragma unroll
  for (int kk = 0; kk < 16; ++kk)
    ATT[(size_t)s * LSP + kk * 256 + tid] = f2bf(v[kk] * inv);
}

// ---- GEMM2: U[s,k] = sum_l ATT[s,l] * A[l,k]  (M=4096, N=Kdim, K=4096) -----
// 128x64 tile, wave -> 32x32 (2x2 frags).  Double-buffered async staging
// (3 async b128 per thread/tile).

__global__ __launch_bounds__(256) void gemm_att_apply_k(const unsigned short* __restrict__ ATT,
                                                        const unsigned short* __restrict__ Ab,
                                                        float* __restrict__ U, int Kdim) {
  __shared__ unsigned short sAtt[2][128 * LDA];  // [m(s)][kk(l)]
  __shared__ unsigned short sB[2][32 * LDB];     // [kk(l)][n(k)]
  const int tid = threadIdx.x, lane = tid & 31, wave = tid >> 5;
  const int wm = wave >> 1, wn = wave & 1;
  const int m0 = blockIdx.x * 128;               // s tile
  const int n0 = blockIdx.y * 64;                // patch-k tile

  const int r  = tid >> 1;
  const int hh = (tid & 1) << 4;
  const unsigned short* gT = ATT + (size_t)(m0 + r) * LSP + hh;
  const int rowoffT = r * LDA + hh;
  const int rb = tid >> 3, cb = (tid & 7) << 3;  // 32 rows x 8 b128 chunks
  const unsigned short* gB = Ab + (size_t)rb * Kdim + n0 + cb;
  const int rowoffB = rb * LDB + cb;

  v8f acc[2][2];
#pragma unroll
  for (int mi = 0; mi < 2; ++mi)
#pragma unroll
    for (int ni = 0; ni < 2; ++ni) acc[mi][ni] = {};

  async_b128(&sAtt[0][rowoffT],     gT);
  async_b128(&sAtt[0][rowoffT + 8], gT + 8);
  async_b128(&sB[0][rowoffB],       gB);

  for (int lt = 0; lt < LSP / 32; ++lt) {
    const int cur = lt & 1, nxt = cur ^ 1;
    if (lt + 1 < LSP / 32) {
      const unsigned short* nT = gT + (size_t)(lt + 1) * 32;
      const unsigned short* nB = gB + (size_t)(lt + 1) * 32 * Kdim;
      async_b128(&sAtt[nxt][rowoffT],     nT);
      async_b128(&sAtt[nxt][rowoffT + 8], nT + 8);
      async_b128(&sB[nxt][rowoffB],       nB);
      WAIT_ASYNC(0x3);
    } else {
      WAIT_ASYNC(0x0);
    }
    __syncthreads();

    v16bf af[2], bfm[2];
#pragma unroll
    for (int mi = 0; mi < 2; ++mi) af[mi] = frag_a(sAtt[cur], wm * 32 + mi * 16, LDA, lane);
#pragma unroll
    for (int ni = 0; ni < 2; ++ni) bfm[ni] = frag_b_cols(sB[cur], wn * 32 + ni * 16, LDB, lane);
#pragma unroll
    for (int mi = 0; mi < 2; ++mi)
#pragma unroll
      for (int ni = 0; ni < 2; ++ni)
        acc[mi][ni] = __builtin_amdgcn_wmma_f32_16x16x32_bf16(
            false, af[mi], false, bfm[ni], (short)0, acc[mi][ni], false, false);
    __syncthreads();
  }
#pragma unroll
  for (int mi = 0; mi < 2; ++mi)
#pragma unroll
    for (int ni = 0; ni < 2; ++ni)
      store_frag(U, (size_t)Kdim, m0 + wm * 32 + mi * 16, n0 + wn * 32 + ni * 16, lane, acc[mi][ni]);
}

// ---- col2im (conv_transpose scatter) + mask blend --------------------------

__global__ __launch_bounds__(256) void col2im_combine_k(const float* __restrict__ U,
                                                        const float* __restrict__ fg,
                                                        const float* __restrict__ mask,
                                                        float* __restrict__ outs_bc,
                                                        int p, int Kdim) {
  const int idx = blockIdx.x * 256 + threadIdx.x;
  if (idx >= NCH * LSP) return;
  const int c = idx >> 12, s = idx & 4095;
  const int y = s >> 6, x = s & 63;
  float rec;
  if (p == 1) {
    rec = U[(size_t)s * Kdim + c];
  } else {
    rec = 0.0f;
#pragma unroll
    for (int i = 0; i < 3; ++i) {
      const int yy = y + 1 - i;
      if (yy < 0 || yy >= 64) continue;
#pragma unroll
      for (int j = 0; j < 3; ++j) {
        const int xx = x + 1 - j;
        if (xx < 0 || xx >= 64) continue;
        rec += U[(size_t)(yy * 64 + xx) * Kdim + c * 9 + i * 3 + j];
      }
    }
  }
  const float m = mask[s];
  outs_bc[(size_t)c * LSP + s] =
      rec * m * (1.0f / (float)(p * p)) + fg[(size_t)c * LSP + s] * (1.0f - m);
}

// ---- epilogue: channel means, SE gate, 1x1 combiner ------------------------

__global__ __launch_bounds__(256) void chan_mean_k(const float* __restrict__ outs,
                                                   float* __restrict__ smean) {
  __shared__ float red[256];
  const int bc = blockIdx.x, tid = threadIdx.x;
  const float* p = outs + (size_t)bc * LSP;
  float s = 0.0f;
  for (int i = tid; i < LSP; i += 256) s += p[i];
  red[tid] = s; __syncthreads();
  for (int st = 128; st > 0; st >>= 1) {
    if (tid < st) red[tid] += red[tid + st];
    __syncthreads();
  }
  if (tid == 0) smean[bc] = red[0] * (1.0f / (float)LSP);
}

__global__ __launch_bounds__(128) void se_gate_k(const float* __restrict__ smean,
                                                 const float* __restrict__ W1,
                                                 const float* __restrict__ b1,
                                                 const float* __restrict__ W2,
                                                 const float* __restrict__ b2,
                                                 float* __restrict__ gate) {
  __shared__ float h[128];
  const int b = blockIdx.x, t = threadIdx.x;
  float a = b1[t];
  for (int j = 0; j < 128; ++j) a += smean[b * 128 + j] * W1[t * 128 + j];
  h[t] = fmaxf(a, 0.0f);
  __syncthreads();
  float a2 = b2[t];
  for (int j = 0; j < 128; ++j) a2 += h[j] * W2[t * 128 + j];
  gate[b * 128 + t] = 1.0f / (1.0f + __expf(-a2));
}

__global__ __launch_bounds__(256) void combine_out_k(const float* __restrict__ outs,
                                                     const float* __restrict__ gate,
                                                     const float* __restrict__ Wc,
                                                     const float* __restrict__ bcv,
                                                     float* __restrict__ out) {
  const int idx = blockIdx.x * 256 + threadIdx.x;   // (b,o,s) flat: 2*64*4096
  const int b = idx >> 18;
  const int o = (idx >> 12) & 63;
  const int s = idx & 4095;
  float acc = bcv[o];
  for (int c = 0; c < 128; ++c)
    acc += outs[((size_t)b * 128 + c) * LSP + s] * gate[b * 128 + c] * Wc[o * 128 + c];
  out[idx] = acc;
}

// ---------------------------------------------------------------------------

extern "C" void kernel_launch(void* const* d_in, const int* in_sizes, int n_in,
                              void* d_out, int out_size, void* d_ws, size_t ws_size,
                              hipStream_t stream) {
  (void)in_sizes; (void)n_in; (void)out_size; (void)ws_size;
  const float* fg   = (const float*)d_in[0];
  const float* mask = (const float*)d_in[1];
  const float* W1   = (const float*)d_in[2];
  const float* b1   = (const float*)d_in[3];
  const float* W2   = (const float*)d_in[4];
  const float* b2   = (const float*)d_in[5];
  const float* Wc   = (const float*)d_in[6];
  const float* bc   = (const float*)d_in[7];
  float* out = (float*)d_out;

  char* ws = (char*)d_ws;
  size_t off = 0;
  float*          ST   = (float*)(ws + off);          off += (size_t)LSP * LSP * 4;   // 64 MB
  unsigned short* ATT  = (unsigned short*)(ws + off); off += (size_t)LSP * LSP * 2;   // 32 MB
  unsigned short* Ab   = (unsigned short*)(ws + off); off += (size_t)LSP * KMAX * 2;
  unsigned short* Fb   = (unsigned short*)(ws + off); off += (size_t)LSP * KMAX * 2;
  float*          U    = (float*)(ws + off);          off += (size_t)LSP * KMAX * 4;
  float*          outs = (float*)(ws + off);          off += (size_t)2 * 128 * LSP * 4;
  float*          smean= (float*)(ws + off);          off += 1024;
  float*          gate = (float*)(ws + off);          off += 1024;

  for (int b = 0; b < 2; ++b) {
    const float* fgS = fg + (size_t)b * NCH * LSP;
    const float* mS  = mask + (size_t)b * LSP;
    for (int br = 0; br < 2; ++br) {
      const int p = (br == 0) ? 1 : 3;
      const int Kdim = NCH * p * p;          // 64 or 576
      build_F_k<<<dim3((LSP * Kdim + 255) / 256), dim3(256), 0, stream>>>(fgS, Fb, p, Kdim);
      build_A_k<<<dim3(LSP), dim3(256), 0, stream>>>(fgS, mS, Ab, p, Kdim);
      gemm_scores_k<<<dim3(32, 32), dim3(256), 0, stream>>>(Fb, Ab, ST, Kdim);
      box_softmax_k<<<dim3(LSP), dim3(256), 0, stream>>>(ST, ATT);
      gemm_att_apply_k<<<dim3(32, Kdim / 64), dim3(256), 0, stream>>>(ATT, Ab, U, Kdim);
      float* outs_bc = outs + ((size_t)b * 128 + br * 64) * LSP;
      col2im_combine_k<<<dim3((NCH * LSP + 255) / 256), dim3(256), 0, stream>>>(
          U, fgS, mS, outs_bc, p, Kdim);
    }
  }
  chan_mean_k<<<dim3(256), dim3(256), 0, stream>>>(outs, smean);
  se_gate_k<<<dim3(2), dim3(128), 0, stream>>>(smean, W1, b1, W2, b2, gate);
  combine_out_k<<<dim3(2048), dim3(256), 0, stream>>>(outs, gate, Wc, bc, out);
}
